// SpotRNAWithTransformer_25220047962437
// MI455X (gfx1250) — matrix operand order repaired and checked by reading, not compiled, not run on hardware
//
#include <hip/hip_runtime.h>
#include <math.h>

typedef __attribute__((ext_vector_type(16))) __bf16 v16bf;
typedef __attribute__((ext_vector_type(8)))  float  v8f;
typedef __attribute__((ext_vector_type(8)))  unsigned short us8;

#define HD   256
#define LSEQ 512

#define WMMA_BF16(a, b, c) \
  __builtin_amdgcn_wmma_f32_16x16x32_bf16(false, (a), false, (b), (short)0, (c), false, false)

__device__ __forceinline__ unsigned short f2bf(float f) {
  union { float f; unsigned u; } x; x.f = f;
  unsigned u = x.u;
  u += 0x7fffu + ((u >> 16) & 1u);          // round-to-nearest-even
  return (unsigned short)(u >> 16);
}
__device__ __forceinline__ float elu1(float x) { return x > 0.f ? x : (expf(x) - 1.f); }

// ---------------------------------------------------------------------------
// bf16 WMMA GEMM. Operands are bf16 (pre-converted). B is always NxK row-major.
//   MODEA: 0 = row-major MxK; 1 = implicit im2col conv1d(k=3) over HD channels
//   EPI:   0 none | 1 +bias | 2 +bias,relu | 3 +bias,+res | 4 BN+ELU | 5 BN+ELU,+res
//   WF32 / WBF16: which output precisions to store (C fp32 / Cb bf16)
// block = 128 threads (4 waves); wave tile = 32(M) x 64(N) = 8 WMMA tiles:
// two A fragments share four B fragments; each B fragment is consumed by its
// two WMMAs immediately after load (low register pressure — no spills; the
// compiler unrolls K-steps and hoists loads with partial s_wait_loadcnt).
// Batched via blockIdx.z -> b=z>>2, h=z&3.
// ---------------------------------------------------------------------------
template<int MODEA, int EPI, int WF32, int WBF16>
__global__ void wmma_gemm(
    const unsigned short* __restrict__ A, const unsigned short* __restrict__ B,
    float* __restrict__ C, unsigned short* __restrict__ Cb,
    int M, int N, int K, int lda, int ldb, int ldc,
    long long sAb, long long sAh, long long sBb, long long sBh,
    long long sCb, long long sCh,
    const float* __restrict__ bias, const float* __restrict__ resid,
    const float* __restrict__ bng, const float* __restrict__ bnb,
    const float* __restrict__ bnm, const float* __restrict__ bnv,
    const float* __restrict__ cb)
{
  const int z  = blockIdx.z;
  const int bb = z >> 2, hh = z & 3;
  const unsigned short* Ap = A + bb * sAb + hh * sAh;
  const unsigned short* Bp = B + bb * sBb + hh * sBh;
  float*          Cp  = C  ? C  + bb * sCb + hh * sCh : nullptr;
  unsigned short* Cbp = Cb ? Cb + bb * sCb + hh * sCh : nullptr;

  const int lane = threadIdx.x & 31;
  const int wave = threadIdx.x >> 5;
  const int g    = lane >> 4;          // half-wave select
  const int lr   = lane & 15;
  const int m_base = blockIdx.y * 128 + wave * 32;
  const int n_base = blockIdx.x * 64;

  // A fragment (16-bit A 16x32, ISA 7.12.2): per lane two contiguous 8-elem
  // bf16 runs at [kb, kb+8) and [kb+16, kb+24), kb = k0 + (g?8:0)
  auto loadA = [&](int mt, int k0) -> v16bf {
    union { v16bf v; us8 h[2]; unsigned short u[16]; } af;
    const int m = m_base + mt * 16 + lr;
    if (MODEA == 0) {
      const unsigned short* arow = Ap + (long long)m * lda + (k0 + (g ? 8 : 0));
      af.h[0] = *(const us8*)(arow);
      af.h[1] = *(const us8*)(arow + 16);
    } else {
      const int t  = k0 >> 8;                       // conv tap
      const int c0 = (k0 & 255) + (g ? 8 : 0);
      const int b  = m >> 9, l = m & 511;
      const int l2 = l + t - 1;
      if ((unsigned)l2 < (unsigned)LSEQ) {
        const unsigned short* arow = Ap + ((long long)((b << 9) + l2)) * HD + c0;
        af.h[0] = *(const us8*)(arow);
        af.h[1] = *(const us8*)(arow + 16);
      } else {
        #pragma unroll
        for (int i = 0; i < 16; ++i) af.u[i] = 0;
      }
    }
    return af.v;
  };
  // B fragment: 32x16 tile, per lane two 8-elem bf16 runs at row n: [kb, kb+8) and [kb+16, kb+24), kb=k0+(g?8:0)
  auto loadB = [&](int nt, int k0) -> v16bf {
    union { v16bf v; us8 h[2]; } bf;
    const unsigned short* brow =
        Bp + (long long)(n_base + nt * 16 + lr) * ldb + (k0 + (g ? 8 : 0));
    bf.h[0] = *(const us8*)(brow);
    bf.h[1] = *(const us8*)(brow + 16);
    return bf.v;
  };

  v8f c00 = {0,0,0,0,0,0,0,0}, c01 = {0,0,0,0,0,0,0,0};
  v8f c02 = {0,0,0,0,0,0,0,0}, c03 = {0,0,0,0,0,0,0,0};
  v8f c10 = {0,0,0,0,0,0,0,0}, c11 = {0,0,0,0,0,0,0,0};
  v8f c12 = {0,0,0,0,0,0,0,0}, c13 = {0,0,0,0,0,0,0,0};

  for (int k0 = 0; k0 < K; k0 += 32) {
    v16bf a0 = loadA(0, k0);
    v16bf a1 = loadA(1, k0);
    {
      v16bf b = loadB(0, k0);
      c00 = WMMA_BF16(a0, b, c00);
      c10 = WMMA_BF16(a1, b, c10);
    }
    {
      v16bf b = loadB(1, k0);
      c01 = WMMA_BF16(a0, b, c01);
      c11 = WMMA_BF16(a1, b, c11);
    }
    {
      v16bf b = loadB(2, k0);
      c02 = WMMA_BF16(a0, b, c02);
      c12 = WMMA_BF16(a1, b, c12);
    }
    {
      v16bf b = loadB(3, k0);
      c03 = WMMA_BF16(a0, b, c03);
      c13 = WMMA_BF16(a1, b, c13);
    }
  }

  // epilogue + store. C layout: lanes 0-15 M=r, lanes 16-31 M=8+r; N=lane&15
  auto store = [&](int mt, int nt, v8f acc) {
    const int n = n_base + nt * 16 + lr;
    float bs = 0.f, sc = 0.f, sh = 0.f;
    if (EPI == 1 || EPI == 2 || EPI == 3) bs = bias[n];
    if (EPI == 4 || EPI == 5) {
      sc = bng[n] * rsqrtf(bnv[n] + 1e-5f);
      sh = (cb[n] - bnm[n]) * sc + bnb[n];
    }
    #pragma unroll
    for (int r = 0; r < 8; ++r) {
      const int m = m_base + mt * 16 + (g ? 8 : 0) + r;
      float v = acc[r];
      if (EPI == 1 || EPI == 2 || EPI == 3) v += bs;
      if (EPI == 2) v = fmaxf(v, 0.f);
      if (EPI == 3) v += resid[(long long)m * ldc + n];
      if (EPI == 4 || EPI == 5) {
        v = v * sc + sh;
        v = elu1(v);
        if (EPI == 5) v += resid[(long long)m * ldc + n];
      }
      if (WF32)  Cp [(long long)m * ldc + n] = v;
      if (WBF16) Cbp[(long long)m * ldc + n] = f2bf(v);
    }
  };
  store(0, 0, c00); store(0, 1, c01); store(0, 2, c02); store(0, 3, c03);
  store(1, 0, c10); store(1, 1, c11); store(1, 2, c12); store(1, 3, c13);
}

// ---------------------------------------------------------------------------
// Conversion / repack kernels (run once per launch)
// ---------------------------------------------------------------------------
__global__ void cvt_kernel(const float* __restrict__ in, unsigned short* __restrict__ out, int n) {
  int i = blockIdx.x * blockDim.x + threadIdx.x;
  if (i < n) out[i] = f2bf(in[i]);
}

// conv_w (R, O=256, I=256, 3) -> bf16 (R, O, k=t*256+c) row-major
__global__ void convpack_kernel(const float* __restrict__ in, unsigned short* __restrict__ out, int total) {
  int idx = blockIdx.x * blockDim.x + threadIdx.x;
  if (idx >= total) return;
  int k = idx % 768;
  int rn = idx / 768;              // r*256 + n
  int t = k >> 8, c = k & 255;
  out[idx] = f2bf(in[((long long)rn * 256 + c) * 3 + t]);
}

// fc1_w (64,1024) -> bf16 (64,512) slice at column offset
__global__ void fc1slice_kernel(const float* __restrict__ in, unsigned short* __restrict__ out, int off) {
  int idx = blockIdx.x * blockDim.x + threadIdx.x;
  if (idx >= 64 * 512) return;
  int n = idx >> 9, k = idx & 511;
  out[idx] = f2bf(in[n * 1024 + off + k]);
}

// V slice of qkv16 -> per (b,head) d-major: vt[z][d][pos], z = b*4+h
__global__ void vt_kernel(const unsigned short* __restrict__ qkv16, unsigned short* __restrict__ vt) {
  int idx = blockIdx.x * blockDim.x + threadIdx.x;
  if (idx >= 16 * 64 * 512) return;
  int pos = idx & 511;
  int d   = (idx >> 9) & 63;
  int zz  = idx >> 15;
  int b = zz >> 2, h = zz & 3;
  vt[idx] = qkv16[((long long)(b * 512 + pos)) * 768 + 512 + h * 64 + d];
}

// ---------------------------------------------------------------------------
// Small supporting kernels
// ---------------------------------------------------------------------------
__global__ void embed_kernel(const float* __restrict__ x, const float* __restrict__ w,
                             const float* __restrict__ b, float* __restrict__ out,
                             unsigned short* __restrict__ out16, int total) {
  int idx = blockIdx.x * blockDim.x + threadIdx.x;
  if (idx >= total) return;
  int h = idx & 255, m = idx >> 8;
  float s = b[h];
  for (int k = 0; k < 4; ++k) s += x[m * 4 + k] * w[h * 4 + k];
  out[idx] = s;
  out16[idx] = f2bf(s);
}

__global__ void posenc_kernel(const float* __restrict__ xin, float* __restrict__ out,
                              unsigned short* __restrict__ out16, int total) {
  int idx = blockIdx.x * blockDim.x + threadIdx.x;
  if (idx >= total) return;
  int h = idx & 255, m = idx >> 8, l = m & 511;
  int i2 = h & ~1;
  float freq = expf(-(float)i2 * (logf(10000.f) / 256.f));
  float v = (float)l * freq;
  float pe = (h & 1) ? cosf(v) : sinf(v);
  float r = xin[idx] + pe;
  out[idx] = r;
  out16[idx] = f2bf(r);
}

__global__ void ln_kernel(const float* __restrict__ in, const float* __restrict__ gg,
                          const float* __restrict__ bb, float* __restrict__ out,
                          unsigned short* __restrict__ out16) {
  __shared__ float sm[256];
  __shared__ float sv[256];
  int row = blockIdx.x, t = threadIdx.x;
  float x = in[(long long)row * 256 + t];
  sm[t] = x; __syncthreads();
  for (int s = 128; s > 0; s >>= 1) { if (t < s) sm[t] += sm[t + s]; __syncthreads(); }
  float mean = sm[0] / 256.f;
  float d = x - mean;
  sv[t] = d * d; __syncthreads();
  for (int s = 128; s > 0; s >>= 1) { if (t < s) sv[t] += sv[t + s]; __syncthreads(); }
  float var = sv[0] / 256.f;
  float r = d * rsqrtf(var + 1e-5f) * gg[t] + bb[t];
  out[(long long)row * 256 + t] = r;
  out16[(long long)row * 256 + t] = f2bf(r);
}

__global__ void softmax_kernel(const float* __restrict__ scores,
                               unsigned short* __restrict__ out16, float scale) {
  __shared__ float red[256];
  long long row = blockIdx.x;
  const float* p = scores + row * 512;
  int t = threadIdx.x;
  float a = p[t] * scale, b = p[t + 256] * scale;
  red[t] = fmaxf(a, b); __syncthreads();
  for (int s = 128; s > 0; s >>= 1) { if (t < s) red[t] = fmaxf(red[t], red[t + s]); __syncthreads(); }
  float mx = red[0]; __syncthreads();
  float ea = expf(a - mx), eb = expf(b - mx);
  red[t] = ea + eb; __syncthreads();
  for (int s = 128; s > 0; s >>= 1) { if (t < s) red[t] += red[t + s]; __syncthreads(); }
  float inv = 1.f / red[0];
  out16[row * 512 + t]       = f2bf(ea * inv);
  out16[row * 512 + t + 256] = f2bf(eb * inv);
}

__global__ void concat_kernel(const float* __restrict__ a, const float* __restrict__ b,
                              unsigned short* __restrict__ out, int total) {
  int idx = blockIdx.x * blockDim.x + threadIdx.x;
  if (idx >= total) return;
  int c = idx & 511, m = idx >> 9;
  float v = (c < 256) ? a[(long long)m * 256 + c] : b[(long long)m * 256 + (c - 256)];
  out[idx] = f2bf(v);
}

// Pairwise head: out[b,i,j] = 0.5*(f(i,j)+f(j,i))
__global__ void pair_kernel(const float* __restrict__ hi, const float* __restrict__ hj,
                            const float* __restrict__ b1, const float* __restrict__ w2,
                            const float* __restrict__ b2, float* __restrict__ out) {
  __shared__ float sHiI[16][64], sHjJ[16][64], sHiJ[16][64], sHjI[16][64];
  int b = blockIdx.z;
  int i0 = blockIdx.y * 16, j0 = blockIdx.x * 16;
  int t = threadIdx.x;
  for (int e = t; e < 16 * 64; e += 256) {
    int r = e >> 6, c = e & 63;
    sHiI[r][c] = hi[((long long)(b * 512 + i0 + r)) * 64 + c];
    sHjJ[r][c] = hj[((long long)(b * 512 + j0 + r)) * 64 + c];
    sHiJ[r][c] = hi[((long long)(b * 512 + j0 + r)) * 64 + c];
    sHjI[r][c] = hj[((long long)(b * 512 + i0 + r)) * 64 + c];
  }
  __syncthreads();
  int ii = t >> 4, jj = t & 15;
  float s1 = 0.f, s2 = 0.f;
  for (int c = 0; c < 64; ++c) {
    float bc = b1[c], wc = w2[c];
    s1 += elu1(sHiI[ii][c] + sHjJ[jj][c] + bc) * wc;
    s2 += elu1(sHiJ[jj][c] + sHjI[ii][c] + bc) * wc;
  }
  out[((long long)b * 512 + (i0 + ii)) * 512 + (j0 + jj)] = 0.5f * (s1 + s2) + b2[0];
}

// ---------------------------------------------------------------------------
// Host orchestration
// ---------------------------------------------------------------------------
template<int MA, int EPI, int WF32, int WBF16>
static inline void run_gemm(hipStream_t stream,
    const unsigned short* A, const unsigned short* B, float* C, unsigned short* Cb,
    int M, int N, int K, int lda, int ldb, int ldc,
    long long sAb, long long sAh, long long sBb, long long sBh,
    long long sCb, long long sCh, int nz,
    const float* bias, const float* res,
    const float* g, const float* bt, const float* mm, const float* vv, const float* cbv)
{
  dim3 grid(N / 64, M / 128, nz), block(128);
  wmma_gemm<MA, EPI, WF32, WBF16><<<grid, block, 0, stream>>>(A, B, C, Cb,
      M, N, K, lda, ldb, ldc, sAb, sAh, sBb, sBh, sCb, sCh, bias, res, g, bt, mm, vv, cbv);
}

extern "C" void kernel_launch(void* const* d_in, const int* in_sizes, int n_in,
                              void* d_out, int out_size, void* d_ws, size_t ws_size,
                              hipStream_t stream) {
  (void)in_sizes; (void)n_in; (void)out_size; (void)ws_size;
  const float* x       = (const float*)d_in[0];
  const float* emb_w   = (const float*)d_in[1];
  const float* emb_b   = (const float*)d_in[2];
  const float* conv_w1 = (const float*)d_in[3];
  const float* conv_b1 = (const float*)d_in[4];
  const float* bn_g1   = (const float*)d_in[5];
  const float* bn_b1   = (const float*)d_in[6];
  const float* bn_m1   = (const float*)d_in[7];
  const float* bn_v1   = (const float*)d_in[8];
  const float* conv_w2 = (const float*)d_in[9];
  const float* conv_b2 = (const float*)d_in[10];
  const float* bn_g2   = (const float*)d_in[11];
  const float* bn_b2   = (const float*)d_in[12];
  const float* bn_m2   = (const float*)d_in[13];
  const float* bn_v2   = (const float*)d_in[14];
  const float* qkv_w   = (const float*)d_in[15];
  const float* qkv_b   = (const float*)d_in[16];
  const float* out_w   = (const float*)d_in[17];
  const float* out_b   = (const float*)d_in[18];
  const float* ln1_g   = (const float*)d_in[19];
  const float* ln1_b   = (const float*)d_in[20];
  const float* ff_w1   = (const float*)d_in[21];
  const float* ff_b1   = (const float*)d_in[22];
  const float* ff_w2   = (const float*)d_in[23];
  const float* ff_b2   = (const float*)d_in[24];
  const float* ln2_g   = (const float*)d_in[25];
  const float* ln2_b   = (const float*)d_in[26];
  const float* fc1_w   = (const float*)d_in[27];
  const float* fc1_b   = (const float*)d_in[28];
  const float* fc2_w   = (const float*)d_in[29];
  const float* fc2_b   = (const float*)d_in[30];
  float* outp = (float*)d_out;

  const int B = 4, L = 512, H = 256, R = 8, NL = 2;
  const int M = B * L;                         // 2048

  // ---- fp32 workspace ----
  float* wsf = (float*)d_ws;
  float* hA  = wsf;  wsf += (size_t)M * H;
  float* hB  = wsf;  wsf += (size_t)M * H;
  float* hC  = wsf;  wsf += (size_t)M * H;
  float* tb  = wsf;  wsf += (size_t)M * H;
  float* sc  = wsf;  wsf += (size_t)B * 4 * L * L;
  float* hib = wsf;  wsf += (size_t)M * 64;
  float* hjb = wsf;  wsf += (size_t)M * 64;

  // ---- bf16 workspace (ushort) ----
  unsigned short* wsu = (unsigned short*)wsf;
  unsigned short* hA16   = wsu;  wsu += (size_t)M * H;
  unsigned short* hB16   = wsu;  wsu += (size_t)M * H;
  unsigned short* hC16   = wsu;  wsu += (size_t)M * H;
  unsigned short* tb16   = wsu;  wsu += (size_t)M * H;
  unsigned short* qkv16  = wsu;  wsu += (size_t)M * 3 * H;
  unsigned short* p16    = wsu;  wsu += (size_t)B * 4 * L * L;
  unsigned short* vt16   = wsu;  wsu += (size_t)16 * 64 * L;
  unsigned short* ao16   = wsu;  wsu += (size_t)M * H;
  unsigned short* ffb16  = wsu;  wsu += (size_t)M * 4 * H;
  unsigned short* xc16   = wsu;  wsu += (size_t)M * 2 * H;
  unsigned short* cw1_16 = wsu;  wsu += (size_t)R * H * 3 * H;
  unsigned short* cw2_16 = wsu;  wsu += (size_t)R * H * 3 * H;
  unsigned short* qkvw16 = wsu;  wsu += (size_t)NL * 3 * H * H;
  unsigned short* outw16 = wsu;  wsu += (size_t)NL * H * H;
  unsigned short* ffw116 = wsu;  wsu += (size_t)NL * 4 * H * H;
  unsigned short* ffw216 = wsu;  wsu += (size_t)NL * H * 4 * H;
  unsigned short* fc1a16 = wsu;  wsu += (size_t)64 * 512;
  unsigned short* fc1b16 = wsu;  wsu += (size_t)64 * 512;

  const long long Z = 0;
  const float* NUL = nullptr;
  auto blk = [](int n) { return dim3((n + 255) / 256); };

  // 0) one-time weight conversion / repack to bf16
  convpack_kernel<<<blk(R*H*3*H), dim3(256), 0, stream>>>(conv_w1, cw1_16, R*H*3*H);
  convpack_kernel<<<blk(R*H*3*H), dim3(256), 0, stream>>>(conv_w2, cw2_16, R*H*3*H);
  cvt_kernel<<<blk(NL*3*H*H), dim3(256), 0, stream>>>(qkv_w, qkvw16, NL*3*H*H);
  cvt_kernel<<<blk(NL*H*H), dim3(256), 0, stream>>>(out_w, outw16, NL*H*H);
  cvt_kernel<<<blk(NL*4*H*H), dim3(256), 0, stream>>>(ff_w1, ffw116, NL*4*H*H);
  cvt_kernel<<<blk(NL*H*4*H), dim3(256), 0, stream>>>(ff_w2, ffw216, NL*H*4*H);
  fc1slice_kernel<<<blk(64*512), dim3(256), 0, stream>>>(fc1_w, fc1a16, 0);
  fc1slice_kernel<<<blk(64*512), dim3(256), 0, stream>>>(fc1_w, fc1b16, 512);

  // 1) embedding
  embed_kernel<<<blk(M*H), dim3(256), 0, stream>>>(x, emb_w, emb_b, hA, hA16, M * H);

  // 2) ResNet1D stack: implicit-im2col bf16 WMMA GEMM, fused BN(eval)+ELU(+res)
  float* cur = hA; unsigned short* cur16 = hA16;
  float* spare = hC; unsigned short* spare16 = hC16;
  for (int i = 0; i < R; ++i) {
    const unsigned short* w1 = cw1_16 + (size_t)i * H * 3 * H;
    const unsigned short* w2 = cw2_16 + (size_t)i * H * 3 * H;
    run_gemm<1, 4, 0, 1>(stream, cur16, w1, nullptr, hB16, M, H, 3 * H, H, 3 * H, H,
                         Z,Z,Z,Z,Z,Z, 1, NUL, NUL,
                         bn_g1 + i*H, bn_b1 + i*H, bn_m1 + i*H, bn_v1 + i*H, conv_b1 + i*H);
    run_gemm<1, 5, 1, 1>(stream, hB16, w2, spare, spare16, M, H, 3 * H, H, 3 * H, H,
                         Z,Z,Z,Z,Z,Z, 1, NUL, cur,
                         bn_g2 + i*H, bn_b2 + i*H, bn_m2 + i*H, bn_v2 + i*H, conv_b2 + i*H);
    float* t2 = cur; cur = spare; spare = t2;
    unsigned short* t3 = cur16; cur16 = spare16; spare16 = t3;
  }
  const float* x_res = cur;  // (B,L,H) fp32

  // 3) positional encoding into transformer stream (fp32 + bf16)
  posenc_kernel<<<blk(M*H), dim3(256), 0, stream>>>(x_res, tb, tb16, M * H);

  // 4) transformer encoder layers
  for (int l = 0; l < NL; ++l) {
    const unsigned short* qw = qkvw16 + (size_t)l * 3 * H * H;
    const float* qb = qkv_b + (size_t)l * 3 * H;
    const unsigned short* ow = outw16 + (size_t)l * H * H;
    const float* ob = out_b + (size_t)l * H;

    // QKV projection (bf16 out only)
    run_gemm<0, 1, 0, 1>(stream, tb16, qw, nullptr, qkv16, M, 3 * H, H, H, H, 3 * H,
                         Z,Z,Z,Z,Z,Z, 1, qb, NUL, NUL, NUL, NUL, NUL, NUL);

    // V -> per-(b,head) d-major transpose for the P*V GEMM
    vt_kernel<<<blk(16*64*L), dim3(256), 0, stream>>>(qkv16, vt16);

    // scores = Q * K^T per (b,head): batched, z = b*4 + h  (fp32 out for softmax)
    run_gemm<0, 0, 1, 0>(stream, qkv16, qkv16 + H, sc, nullptr, L, L, 64, 3 * H, 3 * H, L,
                         (long long)L * 3 * H, 64, (long long)L * 3 * H, 64,
                         (long long)4 * L * L, (long long)L * L, 16,
                         NUL, NUL, NUL, NUL, NUL, NUL, NUL);

    softmax_kernel<<<dim3(B * 4 * L), dim3(256), 0, stream>>>(sc, p16, 0.125f);

    // attn_out = P * V   (B = vt16, NxK row-major; bf16 out only)
    run_gemm<0, 0, 0, 1>(stream, p16, vt16, nullptr, ao16, L, 64, L, L, L, H,
                         (long long)4 * L * L, (long long)L * L,
                         (long long)4 * 64 * L, (long long)64 * L,
                         (long long)L * H, 64, 16,
                         NUL, NUL, NUL, NUL, NUL, NUL, NUL);

    // out projection + residual (fp32), then LayerNorm -> tb (fp32 + bf16)
    run_gemm<0, 3, 1, 0>(stream, ao16, ow, hB, nullptr, M, H, H, H, H, H,
                         Z,Z,Z,Z,Z,Z, 1, ob, tb, NUL, NUL, NUL, NUL, NUL);
    ln_kernel<<<dim3(M), dim3(256), 0, stream>>>(hB, ln1_g + l*H, ln1_b + l*H, tb, tb16);

    // FFN
    const unsigned short* fw1 = ffw116 + (size_t)l * 4 * H * H;
    const float* fb1 = ff_b1 + (size_t)l * 4 * H;
    const unsigned short* fw2 = ffw216 + (size_t)l * H * 4 * H;
    const float* fb2 = ff_b2 + (size_t)l * H;
    run_gemm<0, 2, 0, 1>(stream, tb16, fw1, nullptr, ffb16, M, 4 * H, H, H, H, 4 * H,
                         Z,Z,Z,Z,Z,Z, 1, fb1, NUL, NUL, NUL, NUL, NUL, NUL);
    run_gemm<0, 3, 1, 0>(stream, ffb16, fw2, hB, nullptr, M, H, 4 * H, 4 * H, 4 * H, H,
                         Z,Z,Z,Z,Z,Z, 1, fb2, tb, NUL, NUL, NUL, NUL, NUL);
    ln_kernel<<<dim3(M), dim3(256), 0, stream>>>(hB, ln2_g + l*H, ln2_b + l*H, tb, tb16);
  }

  // 5) pairwise head: xc = [x_res, t] -> bf16 (2048 x 512)
  concat_kernel<<<blk(M*2*H), dim3(256), 0, stream>>>(x_res, tb, xc16, M * 2 * H);

  run_gemm<0, 0, 1, 0>(stream, xc16, fc1a16, hib, nullptr, M, 64, 2 * H, 2 * H, 2 * H, 64,
                       Z,Z,Z,Z,Z,Z, 1, NUL, NUL, NUL, NUL, NUL, NUL, NUL);
  run_gemm<0, 0, 1, 0>(stream, xc16, fc1b16, hjb, nullptr, M, 64, 2 * H, 2 * H, 2 * H, 64,
                       Z,Z,Z,Z,Z,Z, 1, NUL, NUL, NUL, NUL, NUL, NUL, NUL);

  // pair + symmetrize -> d_out
  pair_kernel<<<dim3(L / 16, L / 16, B), dim3(256), 0, stream>>>(hib, hjb, fc1_b, fc2_w, fc2_b, outp);
}
